// BERT_8495445311962
// MI455X (gfx1250) — compile-verified, matrix operand order checked by reading, not run on hardware
//
#include <hip/hip_runtime.h>

typedef __attribute__((ext_vector_type(16))) __bf16 v16bf;
typedef __attribute__((ext_vector_type(8)))  float  v8f;

#define N_NODES 50000
#define F_IN    128
#define NHEAD   8
#define F_OUT   16
#define HF      128          // NHEAD * F_OUT
#define E_EDGES 800000

// Pack two f32 into one dword of two bf16 (truncation) with a single
// v_perm_b32: result bytes = {f1.b3, f1.b2, f0.b3, f0.b2}.
__device__ __forceinline__ unsigned pk_bf16(float f0, float f1) {
  return __builtin_amdgcn_perm(__float_as_uint(f1), __float_as_uint(f0),
                               0x07060302u);
}

// Pack an 8-float run (two float4) into 4 bf16x2 dwords.
__device__ __forceinline__ void pack8(const float4 q0, const float4 q1,
                                      unsigned* u) {
  u[0] = pk_bf16(q0.x, q0.y);
  u[1] = pk_bf16(q0.z, q0.w);
  u[2] = pk_bf16(q1.x, q1.y);
  u[3] = pk_bf16(q1.z, q1.w);
}

union frag16 {
  v16bf v;
  unsigned u[8];
};

// ---------------------------------------------------------------------------
// Kernel 1: bf16 WMMA GEMM. One wave (32 lanes) computes a 16x16 output tile.
// grid.y tiles 0..7 -> proj = x @ Wp.T ; tiles 8..15 -> d_out = x @ Wskip.T
// A fragment: lane L holds row M = L%16; element j covers
//   K = (j/8)*16 + (L/16)*8 + (j%8)   (CDNA5 16-bit A layout).
// B fragment mirrors it with the lane holding output column o (= W row).
// C layout: VGPR v -> M = v + 8*(L/16), N = L%16.
// ---------------------------------------------------------------------------
__global__ __launch_bounds__(32) void gemm_wmma_bf16(
    const float* __restrict__ x, const float* __restrict__ Wp,
    const float* __restrict__ Wskip, float* __restrict__ proj,
    float* __restrict__ skip_out) {
  const int lane = threadIdx.x;
  const int m    = lane & 15;
  const int half = lane >> 4;
  const int mbase = blockIdx.x * 16;
  const int ct    = blockIdx.y;

  const float* W;
  float* outp;
  int obase;
  if (ct < 8) { W = Wp;    outp = proj;     obase = ct * 16; }
  else        { W = Wskip; outp = skip_out; obase = (ct - 8) * 16; }

  const int row  = mbase + m;        // node row this lane holds for A
  const int ocol = obase + m;        // output column (W row) this lane holds for B
  const float* __restrict__ xr = x + (size_t)row  * F_IN;
  const float* __restrict__ wr = W + (size_t)ocol * F_IN;

  __builtin_prefetch(xr, 0, 1);      // global_prefetch_b8
  __builtin_prefetch(wr, 0, 1);

  v8f c = {};
#pragma unroll
  for (int kk = 0; kk < F_IN; kk += 32) {
    frag16 a, b;
    const int b0 = kk + half * 8;        // K run for elements 0..7
    const int b1 = kk + 16 + half * 8;   // K run for elements 8..15
    const float4* xa = (const float4*)(xr + b0);
    const float4* xb = (const float4*)(xr + b1);
    const float4* wa = (const float4*)(wr + b0);
    const float4* wb = (const float4*)(wr + b1);
    pack8(xa[0], xa[1], a.u);        // elements 0..7
    pack8(xb[0], xb[1], a.u + 4);    // elements 8..15
    pack8(wa[0], wa[1], b.u);
    pack8(wb[0], wb[1], b.u + 4);
    c = __builtin_amdgcn_wmma_f32_16x16x32_bf16(
        /*neg_a=*/false, a.v, /*neg_b=*/false, b.v,
        /*c_mod=*/(short)0, c, /*reuse_a=*/false, /*reuse_b=*/false);
  }

#pragma unroll
  for (int v = 0; v < 8; ++v) {
    const int mo = v + half * 8;
    outp[(size_t)(mbase + mo) * HF + obase + m] = c[v];
  }
}

// ---------------------------------------------------------------------------
// Kernel 2: c_tp[h] = sum_f Wt[h*16+f] * a_tp[h,f]  (collapses the [E,128]
// trans tensor to one scalar per head).
// ---------------------------------------------------------------------------
__global__ void ctp_kernel(const float* __restrict__ Wt,
                           const float* __restrict__ a_tp,
                           float* __restrict__ c_tp) {
  const int h = threadIdx.x;
  if (h < NHEAD) {
    float c = 0.f;
#pragma unroll
    for (int f = 0; f < F_OUT; ++f) c += Wt[h * F_OUT + f] * a_tp[h * F_OUT + f];
    c_tp[h] = c;
  }
}

// ---------------------------------------------------------------------------
// Kernel 3: per-node attention scores s_src[n,h], s_trg[n,h].
// ---------------------------------------------------------------------------
__global__ void node_scores(const float* __restrict__ proj,
                            const float* __restrict__ a_src,
                            const float* __restrict__ a_trg,
                            float* __restrict__ s_src,
                            float* __restrict__ s_trg) {
  const int t = blockIdx.x * blockDim.x + threadIdx.x;
  if (t >= N_NODES * NHEAD) return;
  const int n = t >> 3, h = t & 7;
  const float4* __restrict__ p  =
      (const float4*)(proj + (size_t)n * HF + h * F_OUT);
  const float4* __restrict__ as = (const float4*)(a_src + h * F_OUT);
  const float4* __restrict__ at = (const float4*)(a_trg + h * F_OUT);
  float ss = 0.f, st = 0.f;
#pragma unroll
  for (int q = 0; q < 4; ++q) {
    const float4 v = p[q], a = as[q], b = at[q];
    ss += v.x * a.x + v.y * a.y + v.z * a.z + v.w * a.w;
    st += v.x * b.x + v.y * b.y + v.z * b.z + v.w * b.w;
  }
  s_src[t] = ss;
  s_trg[t] = st;
}

// ---------------------------------------------------------------------------
// Kernel 4: zero denom, init global-max slot (encoded uint, 0 == -inf).
// ---------------------------------------------------------------------------
__global__ void init_kernel(float* __restrict__ denom, unsigned* __restrict__ gmax) {
  const int t = blockIdx.x * blockDim.x + threadIdx.x;
  if (t < N_NODES * NHEAD) denom[t] = 0.f;
  if (t == 0) *gmax = 0u;
}

__device__ __forceinline__ unsigned enc_f32(float f) {
  const unsigned u = __float_as_uint(f);
  return (u & 0x80000000u) ? ~u : (u | 0x80000000u);
}

// ---------------------------------------------------------------------------
// Kernel 5: edge scores + leaky relu + global max (block reduce + atomicMax).
// ---------------------------------------------------------------------------
__global__ __launch_bounds__(256) void edge_scores(
    const int* __restrict__ src, const int* __restrict__ trg,
    const float* __restrict__ ep, const float* __restrict__ s_src,
    const float* __restrict__ s_trg, const float* __restrict__ c_tp,
    float* __restrict__ scores, unsigned* __restrict__ gmax) {
  __shared__ unsigned red[256];
  const int t = blockIdx.x * 256 + threadIdx.x;
  unsigned my = 0u;
  if (t < E_EDGES * NHEAD) {
    const int e = t >> 3, h = t & 7;
    float s = s_src[src[e] * NHEAD + h] + s_trg[trg[e] * NHEAD + h] + ep[e] * c_tp[h];
    s = (s > 0.f) ? s : 0.2f * s;   // leaky_relu, slope 0.2
    scores[t] = s;
    my = enc_f32(s);
  }
  red[threadIdx.x] = my;
  __syncthreads();
#pragma unroll
  for (int off = 128; off > 0; off >>= 1) {
    if (threadIdx.x < off) {
      const unsigned o = red[threadIdx.x + off];
      if (o > red[threadIdx.x]) red[threadIdx.x] = o;
    }
    __syncthreads();
  }
  if (threadIdx.x == 0) atomicMax(gmax, red[0]);
}

// ---------------------------------------------------------------------------
// Kernel 6: e = exp(score - gmax); denom[trg,h] += e  (segment sum).
// ---------------------------------------------------------------------------
__global__ void edge_exp(const int* __restrict__ trg, float* __restrict__ scores,
                         const unsigned* __restrict__ gmax,
                         float* __restrict__ denom) {
  const int t = blockIdx.x * blockDim.x + threadIdx.x;
  if (t >= E_EDGES * NHEAD) return;
  const unsigned g = *gmax;
  const float gm = __uint_as_float((g & 0x80000000u) ? (g & 0x7FFFFFFFu) : ~g);
  const int e = t >> 3, h = t & 7;
  const float v = __expf(scores[t] - gm);
  scores[t] = v;
  atomicAdd(&denom[trg[e] * NHEAD + h], v);
}

// ---------------------------------------------------------------------------
// Kernel 7: out[trg, h, :] += (e / (denom+eps)) * proj[src, h, :]
// (d_out already holds the skip projection from kernel 1).
// proj gathers are float4 (b128); proj fits in L2 (25.6 MB << 192 MB).
// ---------------------------------------------------------------------------
__global__ void edge_scatter(const int* __restrict__ src, const int* __restrict__ trg,
                             const float* __restrict__ scores,
                             const float* __restrict__ denom,
                             const float* __restrict__ proj,
                             float* __restrict__ out) {
  const int t = blockIdx.x * blockDim.x + threadIdx.x;
  if (t >= E_EDGES * NHEAD) return;
  const int e = t >> 3, h = t & 7;
  const int s = src[e], g = trg[e];
  const float a = scores[t] / (denom[g * NHEAD + h] + 1e-16f);
  const float4* __restrict__ p =
      (const float4*)(proj + (size_t)s * HF + h * F_OUT);
  float* __restrict__ o = out + (size_t)g * HF + h * F_OUT;
#pragma unroll
  for (int q = 0; q < 4; ++q) {
    const float4 v = p[q];
    atomicAdd(&o[q * 4 + 0], a * v.x);
    atomicAdd(&o[q * 4 + 1], a * v.y);
    atomicAdd(&o[q * 4 + 2], a * v.z);
    atomicAdd(&o[q * 4 + 3], a * v.w);
  }
}

// ---------------------------------------------------------------------------
// Kernel 8: out = elu(out + bias), in place.
// ---------------------------------------------------------------------------
__global__ void finalize(float* __restrict__ out, const float* __restrict__ bias) {
  const int t = blockIdx.x * blockDim.x + threadIdx.x;
  if (t >= N_NODES * HF) return;
  const float v = out[t] + bias[t & (HF - 1)];
  out[t] = (v > 0.f) ? v : (__expf(v) - 1.f);
}

// ---------------------------------------------------------------------------
extern "C" void kernel_launch(void* const* d_in, const int* in_sizes, int n_in,
                              void* d_out, int out_size, void* d_ws, size_t ws_size,
                              hipStream_t stream) {
  const float* x     = (const float*)d_in[0];
  const int*   ei    = (const int*)d_in[1];
  const float* ep    = (const float*)d_in[2];
  const float* Wp    = (const float*)d_in[3];
  const float* Wt    = (const float*)d_in[4];
  const float* a_src = (const float*)d_in[5];
  const float* a_trg = (const float*)d_in[6];
  const float* a_tp  = (const float*)d_in[7];
  const float* Wskip = (const float*)d_in[8];
  const float* bias  = (const float*)d_in[9];
  float* out = (float*)d_out;

  const int* src = ei;
  const int* trg = ei + E_EDGES;

  // workspace layout (floats)
  float* ws     = (float*)d_ws;
  float* proj   = ws;                                     // N*128
  float* scores = proj   + (size_t)N_NODES * HF;          // E*8
  float* s_src  = scores + (size_t)E_EDGES * NHEAD;       // N*8
  float* s_trg  = s_src  + (size_t)N_NODES * NHEAD;       // N*8
  float* denom  = s_trg  + (size_t)N_NODES * NHEAD;       // N*8
  float* c_tp   = denom  + (size_t)N_NODES * NHEAD;       // 8
  unsigned* gmax = (unsigned*)(c_tp + NHEAD);             // 1

  // 1) proj = x@Wp.T (ws), skip = x@Wskip.T (d_out), via bf16 WMMA
  dim3 ggrid(N_NODES / 16, 16);
  gemm_wmma_bf16<<<ggrid, 32, 0, stream>>>(x, Wp, Wskip, proj, out);

  // 2) per-head transition-prob coefficient
  ctp_kernel<<<1, NHEAD, 0, stream>>>(Wt, a_tp, c_tp);

  // 3) per-node source/target scores
  node_scores<<<(N_NODES * NHEAD + 255) / 256, 256, 0, stream>>>(
      proj, a_src, a_trg, s_src, s_trg);

  // 4) init denom + global-max slot
  init_kernel<<<(N_NODES * NHEAD + 255) / 256, 256, 0, stream>>>(denom, gmax);

  // 5) edge scores + leaky relu + global max
  edge_scores<<<(E_EDGES * NHEAD + 255) / 256, 256, 0, stream>>>(
      src, trg, ep, s_src, s_trg, c_tp, scores, gmax);

  // 6) exp + segment-sum denominator
  edge_exp<<<(E_EDGES * NHEAD + 255) / 256, 256, 0, stream>>>(
      trg, scores, gmax, denom);

  // 7) attention-weighted aggregation into d_out (which holds skip)
  edge_scatter<<<(E_EDGES * NHEAD + 255) / 256, 256, 0, stream>>>(
      src, trg, scores, denom, proj, out);

  // 8) bias + ELU
  finalize<<<(N_NODES * HF + 255) / 256, 256, 0, stream>>>(out, bias);
}
